// IsokawaPytorchLayer_70489003262226
// MI455X (gfx1250) — matrix operand order repaired and checked by reading, not compiled, not run on hardware
//
#include <hip/hip_runtime.h>
#include <hip/hip_bf16.h>

// Quaternion fully-connected layer (Isokawa) as a real bf16 WMMA GEMM:
//   C(256x8192) = Xflat(256x8192) @ W_real^T,  W_real[4n+k, 4m+j] = sign(k,j) * W[n,m,j^k]
// then out = sigmoid(C - theta).
//
// Per wave: 128x16 output strip = 8 WMMAs per on-the-fly-expanded B fragment.
// Ping-pong LDS A-tiles (one barrier per k-step), software-pipelined global
// loads, and an explicitly interleaved 2-deep A-fragment ds_load pipeline
// (distinct named buffers so the allocator cannot coalesce them).

typedef __attribute__((ext_vector_type(16))) __bf16 v16bf;
typedef __attribute__((ext_vector_type(8)))  float  v8f;

#define BATCH 256
#define MQ    2048
#define NQ    2048
#define KDIM  (4 * MQ)   // 8192
#define CDIM  (4 * NQ)   // 8192
#define ROWS_WG 128      // batch rows per workgroup (8 row-tiles per wave)
#define LDS_U   2048     // uints per LDS buffer: 128 rows * 32 bf16 / 2

union AFrag  { v16bf v; uint4 q[2]; };
union BFragU { v16bf v; unsigned u[8]; };

static __device__ __forceinline__ unsigned pk_bf16(float lo, float hi) {
    unsigned short a = __builtin_bit_cast(unsigned short, (__bf16)lo);
    unsigned short b = __builtin_bit_cast(unsigned short, (__bf16)hi);
    return (unsigned)a | ((unsigned)b << 16);
}

// pack 16 f32 (4 float4) -> 16 bf16 and store as two uint4 into LDS
static __device__ __forceinline__ void store_half(unsigned* dst, const float4* r) {
    uint4 a, b;
    a.x = pk_bf16(r[0].x, r[0].y); a.y = pk_bf16(r[0].z, r[0].w);
    a.z = pk_bf16(r[1].x, r[1].y); a.w = pk_bf16(r[1].z, r[1].w);
    b.x = pk_bf16(r[2].x, r[2].y); b.y = pk_bf16(r[2].z, r[2].w);
    b.z = pk_bf16(r[3].x, r[3].y); b.w = pk_bf16(r[3].z, r[3].w);
    ((uint4*)dst)[0] = a;
    ((uint4*)dst)[1] = b;
}

// Expand 4 quaternions into 16 bf16 B-fragment elements:
// element (mm, j) = sign(kc,j) * q_{m0+mm}[kc ^ j]
// Built as 8 packed 32-bit lanes -> v_cvt_pk_bf16_f32 pairs.
static __device__ __forceinline__ v16bf expand_b(const float4* wq, bool b0, bool b1,
                                                 unsigned sg1, unsigned sg2, unsigned sg3) {
    BFragU bf;
    #pragma unroll
    for (int mm = 0; mm < 4; ++mm) {
        float4 qv = wq[mm];
        float x01  = b0 ? qv.y : qv.x;   // q[b0]
        float x01b = b0 ? qv.x : qv.y;   // q[!b0]
        float x23  = b0 ? qv.w : qv.z;   // q[2+b0]
        float x23b = b0 ? qv.z : qv.w;   // q[2+!b0]
        float e0 = b1 ? x23  : x01;      // q[kc]
        float e1 = b1 ? x23b : x01b;     // q[kc^1]
        float e2 = b1 ? x01  : x23;      // q[kc^2]
        float e3 = b1 ? x01b : x23b;     // q[kc^3]
        float l1 = __uint_as_float(__float_as_uint(e1) ^ sg1);
        float l2 = __uint_as_float(__float_as_uint(e2) ^ sg2);
        float l3 = __uint_as_float(__float_as_uint(e3) ^ sg3);
        bf.u[mm * 2 + 0] = pk_bf16(e0, l1);
        bf.u[mm * 2 + 1] = pk_bf16(l2, l3);
    }
    return bf.v;
}

static __device__ __forceinline__ void load_afrag(AFrag& a, const unsigned* abase, int t) {
    const unsigned* sp = abase + t * 256;           // 16 rows * 16 uints per tile
    a.q[0] = ((const uint4*)sp)[0];                 // K = kb .. kb+7
    a.q[1] = ((const uint4*)(sp + 4))[0];           // K = kb+8 .. kb+15
}

__global__ __launch_bounds__(128) void isokawa_wmma_kernel(
    const float* __restrict__ x,      // (256, 2048, 4)  -> (256, 8192)
    const float* __restrict__ W,      // (2048, 2048, 4)
    const float* __restrict__ theta,  // (2048, 4)       -> (8192)
    float* __restrict__ out)          // (256, 2048, 4)  -> (256, 8192)
{
    // ping-pong A-tile staging: 2 x (128 rows x 32 K bf16) = 16 KB
    __shared__ __align__(16) unsigned sA[2 * LDS_U];

    const int tid  = threadIdx.x;
    const int wave = tid >> 5;
    const int lane = tid & 31;
    const int l15  = lane & 15;
    const int lhi  = lane >> 4;          // 0: lanes 0-15, 1: lanes 16-31

    const int brow0 = blockIdx.y * ROWS_WG;
    const int col0  = blockIdx.x * 64 + wave * 16;

    // ---- per-lane B-matrix column mapping (fixed for whole k-loop) ----
    const int c  = col0 + l15;   // output column = 4n + kc
    const int n  = c >> 2;
    const int kc = c & 3;
    const bool b0 = (kc & 1) != 0;
    const bool b1 = (kc & 2) != 0;
    const unsigned NEGM = 0x80000000u;
    const unsigned sg1 = (kc == 0 || kc == 3) ? NEGM : 0u;
    const unsigned sg2 = (kc < 2)             ? NEGM : 0u;
    const unsigned sg3 = ((kc & 1) == 0)      ? NEGM : 0u;

    // W quads for k-chunk kk live at float offset (kk + lhi*16) in row n
    const float* __restrict__ wbase = W + (size_t)n * (MQ * 4) + lhi * 16;

    // ---- A-tile fill assignment: thread -> (row, 16-K half), 2 units ----
    const int frow  = tid >> 1;   // 0..63
    const int fhalf = tid & 1;
    const float* __restrict__ xf0 = x + (size_t)(brow0 + frow) * KDIM + fhalf * 16;
    const float* __restrict__ xf1 = xf0 + (size_t)64 * KDIM;     // rows 64..127
    const int sdoff = frow * 16 + fhalf * 8;  // uint offset of unit0 (unit1: +1024)

    v8f zero = {};
    v8f acc[8];
    #pragma unroll
    for (int t = 0; t < 8; ++t) acc[t] = zero;

    // ---- prologue: preload W quads (kk=0), stage x tile 0 into buf0 ----
    float4 wqA[4], wqB[4];
    #pragma unroll
    for (int m = 0; m < 4; ++m) wqA[m] = *(const float4*)(wbase + m * 4);
    {
        float4 r[8];
        #pragma unroll
        for (int u = 0; u < 4; ++u) r[u]     = ((const float4*)xf0)[u];
        #pragma unroll
        for (int u = 0; u < 4; ++u) r[4 + u] = ((const float4*)xf1)[u];
        store_half(sA + sdoff, r);
        store_half(sA + sdoff + 1024, r + 4);
    }

    // one k-step: read buf[p], write next tile into buf[p^1]
    auto body = [&](int kk, const float4 (&wqc)[4], float4 (&wqn)[4], int p) {
        const int knext = (kk + 32 < KDIM) ? (kk + 32) : 0;  // wrapped dummy on last
        __syncthreads();                                     // buf[p] ready

        // issue next-tile x loads and next W quads (consumed after the WMMAs)
        float4 r[8];
        #pragma unroll
        for (int u = 0; u < 4; ++u) r[u]     = ((const float4*)(xf0 + knext))[u];
        #pragma unroll
        for (int u = 0; u < 4; ++u) r[4 + u] = ((const float4*)(xf1 + knext))[u];
        #pragma unroll
        for (int m = 0; m < 4; ++m) wqn[m] = *(const float4*)(wbase + knext + m * 4);

        // A-fragment pipeline: issue first two tile loads now, hide their LDS
        // latency under the B-expansion VALU block below.
        // A k order used here (matches B): row M=l15 holds K kb..kb+15, kb=16*lhi
        const unsigned* abase = sA + p * LDS_U + l15 * 16 + (lhi << 3);
        AFrag afE, afO;                       // distinct buffers: do not coalesce
        load_afrag(afE, abase, 0);
        load_afrag(afO, abase, 1);

        // current B fragment (32x16 bf16): lane = column, lanes 16-31 = K 16..31
        v16bf bfrag = expand_b(wqc, b0, b1, sg1, sg2, sg3);

        // explicit 2-deep interleave: wmma(t) then load(t+2)
        acc[0] = __builtin_amdgcn_wmma_f32_16x16x32_bf16(false, afE.v, false, bfrag, (short)0, acc[0], false, false);
        load_afrag(afE, abase, 2);
        acc[1] = __builtin_amdgcn_wmma_f32_16x16x32_bf16(false, afO.v, false, bfrag, (short)0, acc[1], false, false);
        load_afrag(afO, abase, 3);
        acc[2] = __builtin_amdgcn_wmma_f32_16x16x32_bf16(false, afE.v, false, bfrag, (short)0, acc[2], false, false);
        load_afrag(afE, abase, 4);
        acc[3] = __builtin_amdgcn_wmma_f32_16x16x32_bf16(false, afO.v, false, bfrag, (short)0, acc[3], false, false);
        load_afrag(afO, abase, 5);
        acc[4] = __builtin_amdgcn_wmma_f32_16x16x32_bf16(false, afE.v, false, bfrag, (short)0, acc[4], false, false);
        load_afrag(afE, abase, 6);
        acc[5] = __builtin_amdgcn_wmma_f32_16x16x32_bf16(false, afO.v, false, bfrag, (short)0, acc[5], false, false);
        load_afrag(afO, abase, 7);
        acc[6] = __builtin_amdgcn_wmma_f32_16x16x32_bf16(false, afE.v, false, bfrag, (short)0, acc[6], false, false);
        acc[7] = __builtin_amdgcn_wmma_f32_16x16x32_bf16(false, afO.v, false, bfrag, (short)0, acc[7], false, false);

        // stage next tile into the other buffer (visible after next barrier)
        unsigned* obuf = sA + (p ^ 1) * LDS_U;
        store_half(obuf + sdoff, r);
        store_half(obuf + sdoff + 1024, r + 4);
    };

    // 256 k-steps, unrolled by 2 to alternate W-quad register sets (no copies)
    int p = 0;
    for (int kk = 0; kk < KDIM; kk += 64) {
        body(kk,      wqA, wqB, p);
        body(kk + 32, wqB, wqA, p ^ 1);
    }

    // ---- epilogue: subtract theta, sigmoid (fast rcp), store ----
    // C/D layout: lane = column N=l15, VGPR g = row M = g + 8*lhi.
    const float th = theta[c];
    const int rbase = brow0 + (lhi << 3);
    #pragma unroll
    for (int t = 0; t < 8; ++t) {
        #pragma unroll
        for (int g = 0; g < 8; ++g) {
            const int b = rbase + t * 16 + g;
            const float sv = acc[t][g] - th;
            out[(size_t)b * CDIM + c] = __builtin_amdgcn_rcpf(1.0f + __expf(-sv));
        }
    }
}

extern "C" void kernel_launch(void* const* d_in, const int* in_sizes, int n_in,
                              void* d_out, int out_size, void* d_ws, size_t ws_size,
                              hipStream_t stream) {
    const float* x     = (const float*)d_in[0];  // x_batch (256,2048,4)
    const float* W     = (const float*)d_in[1];  // W_q     (2048,2048,4)
    const float* theta = (const float*)d_in[2];  // theta_q (2048,4)
    float* out = (float*)d_out;                  // (256,2048,4)

    dim3 grid(CDIM / 64, BATCH / ROWS_WG);       // (128, 2)
    isokawa_wmma_kernel<<<grid, 128, 0, stream>>>(x, W, theta, out);
}
